// hccexplorer_graph_37125697307059
// MI455X (gfx1250) — matrix operand order repaired and checked by reading, not compiled, not run on hardware
//
#include <hip/hip_runtime.h>
#include <stdint.h>

#define NROWS 100000
#define DIN   896
#define HID   256
#define NLAYERS 2
#define ALPHA 0.5f
#define EPS   1e-5f

typedef __attribute__((ext_vector_type(16))) __bf16 v16bf;
typedef __attribute__((ext_vector_type(8)))  float  v8f;

// ---------- bf16 <-> f32 ----------
// Native RNE convert: lowers to v_cvt_pk_bf16_f32 / v_cvt_bf16_f32 on gfx1250
__device__ __forceinline__ unsigned short f2bf(float f) {
  return __builtin_bit_cast(unsigned short, (__bf16)f);
}
__device__ __forceinline__ unsigned int f2bf_pk(float lo, float hi) {
  return (unsigned int)f2bf(lo) | ((unsigned int)f2bf(hi) << 16);
}
__device__ __forceinline__ float bf2f(unsigned short h) {
  return __uint_as_float(((unsigned int)h) << 16);
}

union FragU { uint4 u[2]; v16bf v; };

// A-matrix 16x32 bf16 fragment (row-major source, ld elements per row):
// lanes 0-15: K = {0..7, 16..23}; lanes 16-31: K = {8..15, 24..31}
__device__ __forceinline__ v16bf load_frag_a_g(const unsigned short* __restrict__ a, int row,
                                               size_t ld, int k0, int half) {
  const unsigned short* p = a + (size_t)row * ld + k0;
  FragU t;
  t.u[0] = *(const uint4*)(p + half * 8);
  t.u[1] = *(const uint4*)(p + 16 + half * 8);
  return t.v;
}
// Same pattern but from an LDS-staged tile As[row][40]
__device__ __forceinline__ v16bf load_frag_a_lds(const unsigned short* rowbase, int half) {
  FragU t;
  t.u[0] = *(const uint4*)(rowbase + half * 8);
  t.u[1] = *(const uint4*)(rowbase + 16 + half * 8);
  return t.v;
}
// B-matrix 32x16 bf16 fragment from a TRANSPOSED (N-major) source:
// lanes 0-15 = col, K = {0..7, 16..23}; lanes 16-31 = col, K = {8..15, 24..31}
__device__ __forceinline__ v16bf load_frag_b_g(const unsigned short* __restrict__ wT, int col,
                                               size_t ld, int k0, int half) {
  const unsigned short* p = wT + (size_t)col * ld + k0;
  FragU t;
  t.u[0] = *(const uint4*)(p + half * 8);
  t.u[1] = *(const uint4*)(p + 16 + half * 8);
  return t.v;
}

__device__ __forceinline__ float wave_sum(float v) {
#pragma unroll
  for (int m = 16; m >= 1; m >>= 1) v += __shfl_xor(v, m, 32);
  return v;
}

// ---------------- weight conversion fp32 -> transposed bf16 ----------------
__global__ void k_convert(const float* __restrict__ fcw, const float* __restrict__ Wq,
                          const float* __restrict__ Wk, const float* __restrict__ Wv,
                          unsigned short* __restrict__ fcwT, unsigned short* __restrict__ wqT,
                          unsigned short* __restrict__ wkT, unsigned short* __restrict__ wvT) {
  int i = blockIdx.x * 256 + threadIdx.x;
  if (i < DIN * HID) {  // fcw[k][j] -> fcwT[j][k]
    int k = i / HID, j = i % HID;
    fcwT[(size_t)j * DIN + k] = f2bf(fcw[i]);
  }
  if (i < NLAYERS * HID * HID) {  // W[l][k][j] -> WT[l][j][k]
    int l = i / (HID * HID), r = i % (HID * HID);
    int k = r / HID, j = r % HID;
    size_t o = (size_t)l * HID * HID + (size_t)j * HID + k;
    wqT[o] = f2bf(Wq[i]);
    wkT[o] = f2bf(Wk[i]);
    wvT[o] = f2bf(Wv[i]);
  }
}

// ---------------- FC GEMM + bias + LayerNorm + ReLU ----------------
// block = 256 threads (8 waves), tile = 32 rows x 256 cols, K = 896
__global__ __launch_bounds__(256)
void k_fc_ln_relu(const float* __restrict__ x, const unsigned short* __restrict__ wT,
                  const float* __restrict__ bias, const float* __restrict__ g,
                  const float* __restrict__ beta, float* __restrict__ outf,
                  unsigned short* __restrict__ outbf) {
  __shared__ __align__(16) unsigned short As[32][40];
  __shared__ float Cs[32][257];
  const int t = threadIdx.x, lane = t & 31, wave = t >> 5;
  const int half = lane >> 4, lr = lane & 15;
  const int row0 = blockIdx.x * 32;
  v8f acc[2][2];
#pragma unroll
  for (int mi = 0; mi < 2; ++mi)
#pragma unroll
    for (int nj = 0; nj < 2; ++nj) { v8f z = {0.f,0.f,0.f,0.f,0.f,0.f,0.f,0.f}; acc[mi][nj] = z; }

  for (int k0 = 0; k0 < DIN; k0 += 32) {
    {  // stage A tile: 32x32 fp32 -> bf16 (native packed converts)
      int r = t >> 3, c = (t & 7) * 4;
      float4 f = *(const float4*)(x + (size_t)(row0 + r) * DIN + k0 + c);
      *(uint2*)&As[r][c] = make_uint2(f2bf_pk(f.x, f.y), f2bf_pk(f.z, f.w));
      if (k0 + 32 < DIN) __builtin_prefetch(x + (size_t)(row0 + r) * DIN + k0 + 32 + c, 0, 1);
    }
    __syncthreads();
    v16bf bfr[2];
#pragma unroll
    for (int nj = 0; nj < 2; ++nj)
      bfr[nj] = load_frag_b_g(wT, (wave * 2 + nj) * 16 + lr, DIN, k0, half);
#pragma unroll
    for (int mi = 0; mi < 2; ++mi) {
      v16bf a = load_frag_a_lds(&As[mi * 16 + lr][0], half);
#pragma unroll
      for (int nj = 0; nj < 2; ++nj)
        acc[mi][nj] = __builtin_amdgcn_wmma_f32_16x16x32_bf16(false, a, false, bfr[nj], (short)0,
                                                              acc[mi][nj], false, false);
    }
    __syncthreads();
  }
  const int col0 = wave * 32;
#pragma unroll
  for (int mi = 0; mi < 2; ++mi)
#pragma unroll
    for (int nj = 0; nj < 2; ++nj) {
      int c = col0 + nj * 16 + lr;
      int rbase = mi * 16 + half * 8;
#pragma unroll
      for (int r = 0; r < 8; ++r) Cs[rbase + r][c] = acc[mi][nj][r] + bias[c];
    }
  __syncthreads();
  for (int rr = 0; rr < 4; ++rr) {
    int r = wave * 4 + rr, row = row0 + r;
    float vals[8], s = 0.f, s2 = 0.f;
#pragma unroll
    for (int j = 0; j < 8; ++j) {
      float v = Cs[r][lane * 8 + j];
      vals[j] = v; s += v; s2 += v * v;
    }
    s = wave_sum(s); s2 = wave_sum(s2);
    float mean = s * (1.f / HID);
    float var  = s2 * (1.f / HID) - mean * mean;
    float rstd = rsqrtf(var + EPS);
#pragma unroll
    for (int j = 0; j < 8; j += 2) {
      int c = lane * 8 + j;
      float v0 = fmaxf((vals[j]     - mean) * rstd * g[c]     + beta[c],     0.f);
      float v1 = fmaxf((vals[j + 1] - mean) * rstd * g[c + 1] + beta[c + 1], 0.f);
      outf[(size_t)row * HID + c]     = v0;
      outf[(size_t)row * HID + c + 1] = v1;
      *(unsigned int*)(outbf + (size_t)row * HID + c) = f2bf_pk(v0, v1);
    }
  }
}

// ---------------- QKV GEMM: out = h @ W + b (direct-global fragments, no GEMM LDS) ------
// optional: row-major bf16 out, transposed bf16 out, Frobenius sumsq, column sums
__global__ __launch_bounds__(256)
void k_qkv(const unsigned short* __restrict__ hbf, const unsigned short* __restrict__ wT,
           const float* __restrict__ bias, unsigned short* __restrict__ out_rm,
           unsigned short* __restrict__ out_tr, float* __restrict__ scal, int ssq_idx,
           float* __restrict__ ksum, int do_colsum) {
  __shared__ float Cs[32][257];
  __shared__ float wsum[8];
  const int t = threadIdx.x, lane = t & 31, wave = t >> 5;
  const int half = lane >> 4, lr = lane & 15;
  const int row0 = blockIdx.x * 32;
  v8f acc[2][2];
#pragma unroll
  for (int mi = 0; mi < 2; ++mi)
#pragma unroll
    for (int nj = 0; nj < 2; ++nj) { v8f z = {0.f,0.f,0.f,0.f,0.f,0.f,0.f,0.f}; acc[mi][nj] = z; }

  for (int k0 = 0; k0 < HID; k0 += 32) {
    v16bf bfr[2];
#pragma unroll
    for (int nj = 0; nj < 2; ++nj)
      bfr[nj] = load_frag_b_g(wT, (wave * 2 + nj) * 16 + lr, HID, k0, half);
#pragma unroll
    for (int mi = 0; mi < 2; ++mi) {
      v16bf a = load_frag_a_g(hbf, row0 + mi * 16 + lr, HID, k0, half);
#pragma unroll
      for (int nj = 0; nj < 2; ++nj)
        acc[mi][nj] = __builtin_amdgcn_wmma_f32_16x16x32_bf16(false, a, false, bfr[nj], (short)0,
                                                              acc[mi][nj], false, false);
    }
  }
  const int col0 = wave * 32;
#pragma unroll
  for (int mi = 0; mi < 2; ++mi)
#pragma unroll
    for (int nj = 0; nj < 2; ++nj) {
      int c = col0 + nj * 16 + lr;
      int rbase = mi * 16 + half * 8;
#pragma unroll
      for (int r = 0; r < 8; ++r) Cs[rbase + r][c] = acc[mi][nj][r] + bias[c];
    }
  __syncthreads();
  // row-major output (packed u32 stores) + Frobenius sum of squares
  float ssq = 0.f;
  {
    int r = t >> 3, cb = (t & 7) * 32;
#pragma unroll
    for (int j = 0; j < 32; j += 2) {
      float v0 = Cs[r][cb + j], v1 = Cs[r][cb + j + 1];
      if (out_rm)
        *(unsigned int*)(out_rm + (size_t)(row0 + r) * HID + cb + j) = f2bf_pk(v0, v1);
      ssq += v0 * v0 + v1 * v1;
    }
  }
  // transposed output: thread t owns column t -> contiguous 64B run in out_tr
  if (out_tr) {
    int c = t;
#pragma unroll
    for (int r = 0; r < 32; r += 2)
      *(unsigned int*)(out_tr + (size_t)c * NROWS + row0 + r) = f2bf_pk(Cs[r][c], Cs[r + 1][c]);
  }
  if (ssq_idx >= 0) {
    float w = wave_sum(ssq);
    if (lane == 0) wsum[wave] = w;
    __syncthreads();
    if (t == 0) {
      float tot = 0.f;
#pragma unroll
      for (int i = 0; i < 8; ++i) tot += wsum[i];
      atomicAdd(&scal[ssq_idx], tot);
    }
  }
  if (do_colsum) {
    float cs = 0.f;
#pragma unroll
    for (int r2 = 0; r2 < 32; ++r2) cs += Cs[r2][t];
    atomicAdd(&ksum[t], cs);
  }
}

// ---------------- KV = K^T V  (split-K over nodes, fp32 atomics, NO LDS) ----------------
// A = kT [HID][NROWS], B = vT [HID][NROWS]; all chunk boundaries are multiples of 32.
__global__ __launch_bounds__(256)
void k_kv(const unsigned short* __restrict__ kT, const unsigned short* __restrict__ vT,
          float* __restrict__ kv, int nTotal, int chunkRows) {
  const int t = threadIdx.x, lane = t & 31, wave = t >> 5;
  const int half = lane >> 4, lr = lane & 15;
  const int m0 = blockIdx.x * 64;
  const int nbase0 = blockIdx.y * chunkRows;
  int nend = nbase0 + chunkRows; if (nend > nTotal) nend = nTotal;
  v8f acc[4][2];
#pragma unroll
  for (int mi = 0; mi < 4; ++mi)
#pragma unroll
    for (int nj = 0; nj < 2; ++nj) { v8f z = {0.f,0.f,0.f,0.f,0.f,0.f,0.f,0.f}; acc[mi][nj] = z; }

  for (int nb = nbase0; nb < nend; nb += 32) {
    v16bf bfr[2];
#pragma unroll
    for (int nj = 0; nj < 2; ++nj)
      bfr[nj] = load_frag_b_g(vT, (wave * 2 + nj) * 16 + lr, NROWS, nb, half);
    if (nb + 32 < nend)
      __builtin_prefetch(vT + (size_t)((wave * 2) * 16 + lr) * NROWS + nb + 32, 0, 1);
#pragma unroll
    for (int mi = 0; mi < 4; ++mi) {
      v16bf a = load_frag_a_g(kT, m0 + mi * 16 + lr, NROWS, nb, half);
#pragma unroll
      for (int nj = 0; nj < 2; ++nj)
        acc[mi][nj] = __builtin_amdgcn_wmma_f32_16x16x32_bf16(false, a, false, bfr[nj], (short)0,
                                                              acc[mi][nj], false, false);
    }
  }
#pragma unroll
  for (int mi = 0; mi < 4; ++mi)
#pragma unroll
    for (int nj = 0; nj < 2; ++nj) {
      int d = (wave * 2 + nj) * 16 + lr;
      int mb = m0 + mi * 16 + half * 8;
#pragma unroll
      for (int r = 0; r < 8; ++r) atomicAdd(&kv[(size_t)(mb + r) * HID + d], acc[mi][nj][r]);
    }
}

// ------- scale KV by 1/(||q|| ||k||), convert to bf16, store TRANSPOSED [d][m] -------
__global__ void k_kv_scale(const float* __restrict__ kv, const float* __restrict__ scal,
                           unsigned short* __restrict__ kvbT) {
  int i = blockIdx.x * 256 + threadIdx.x;   // i = d*HID + m
  int d = i >> 8, m = i & 255;
  float invs = rsqrtf(scal[0]) * rsqrtf(scal[1]);
  kvbT[i] = f2bf(kv[(size_t)m * HID + d] * invs);
}

// -------- attention numerator GEMM + denom + residual + LayerNorm (direct-global) --------
__global__ __launch_bounds__(256)
void k_attn(const unsigned short* __restrict__ qbf, const unsigned short* __restrict__ kvbT,
            const float* __restrict__ ksum, const float* __restrict__ scal,
            const unsigned short* __restrict__ vbf, const float* prev,
            const float* __restrict__ g, const float* __restrict__ beta,
            float* outf, unsigned short* outbf) {
  __shared__ float Cs[32][257];
  const int t = threadIdx.x, lane = t & 31, wave = t >> 5;
  const int half = lane >> 4, lr = lane & 15;
  const int row0 = blockIdx.x * 32;
  v8f acc[2][2];
#pragma unroll
  for (int mi = 0; mi < 2; ++mi)
#pragma unroll
    for (int nj = 0; nj < 2; ++nj) { v8f z = {0.f,0.f,0.f,0.f,0.f,0.f,0.f,0.f}; acc[mi][nj] = z; }

  for (int k0 = 0; k0 < HID; k0 += 32) {
    v16bf bfr[2];
#pragma unroll
    for (int nj = 0; nj < 2; ++nj)
      bfr[nj] = load_frag_b_g(kvbT, (wave * 2 + nj) * 16 + lr, HID, k0, half);
#pragma unroll
    for (int mi = 0; mi < 2; ++mi) {
      v16bf a = load_frag_a_g(qbf, row0 + mi * 16 + lr, HID, k0, half);
#pragma unroll
      for (int nj = 0; nj < 2; ++nj)
        acc[mi][nj] = __builtin_amdgcn_wmma_f32_16x16x32_bf16(false, a, false, bfr[nj], (short)0,
                                                              acc[mi][nj], false, false);
    }
  }
  const int col0 = wave * 32;
#pragma unroll
  for (int mi = 0; mi < 2; ++mi)
#pragma unroll
    for (int nj = 0; nj < 2; ++nj) {
      int c = col0 + nj * 16 + lr;
      int rbase = mi * 16 + half * 8;
#pragma unroll
      for (int r = 0; r < 8; ++r) Cs[rbase + r][c] = acc[mi][nj][r];
    }
  __syncthreads();
  const float invs = rsqrtf(scal[0]) * rsqrtf(scal[1]);
  const float nf = (float)NROWS;
  for (int rr = 0; rr < 4; ++rr) {
    int r = wave * 4 + rr, row = row0 + r;
    float dp = 0.f;
#pragma unroll
    for (int j = 0; j < 8; ++j) {
      int c = lane * 8 + j;
      dp += bf2f(qbf[(size_t)row * HID + c]) * ksum[c];
    }
    dp = wave_sum(dp);
    float rden = 1.0f / (dp * invs + nf);
    float hv[8], s = 0.f, s2 = 0.f;
#pragma unroll
    for (int j = 0; j < 8; ++j) {
      int c = lane * 8 + j;
      float num = Cs[r][c] + nf * bf2f(vbf[(size_t)row * HID + c]);
      float a = num * rden;
      float h = ALPHA * a + (1.f - ALPHA) * prev[(size_t)row * HID + c];
      hv[j] = h; s += h; s2 += h * h;
    }
    s = wave_sum(s); s2 = wave_sum(s2);
    float mean = s * (1.f / HID);
    float var  = s2 * (1.f / HID) - mean * mean;
    float rstd = rsqrtf(var + EPS);
#pragma unroll
    for (int j = 0; j < 8; j += 2) {
      int c = lane * 8 + j;
      float o0 = (hv[j]     - mean) * rstd * g[c]     + beta[c];
      float o1 = (hv[j + 1] - mean) * rstd * g[c + 1] + beta[c + 1];
      outf[(size_t)row * HID + c]     = o0;
      outf[(size_t)row * HID + c + 1] = o1;
      if (outbf)
        *(unsigned int*)(outbf + (size_t)row * HID + c) = f2bf_pk(o0, o1);
    }
  }
}

extern "C" void kernel_launch(void* const* d_in, const int* in_sizes, int n_in,
                              void* d_out, int out_size, void* d_ws, size_t ws_size,
                              hipStream_t stream) {
  (void)in_sizes; (void)n_in; (void)out_size; (void)ws_size;
  const float* x    = (const float*)d_in[0];
  // d_in[1] = edge_index (unused by the linear-attention reference)
  const float* fc_w = (const float*)d_in[2];
  const float* fc_b = (const float*)d_in[3];
  const float* Wq   = (const float*)d_in[4];
  const float* Wk   = (const float*)d_in[5];
  const float* Wv   = (const float*)d_in[6];
  const float* bq   = (const float*)d_in[7];
  const float* bk   = (const float*)d_in[8];
  const float* bv   = (const float*)d_in[9];
  const float* ln_g = (const float*)d_in[10];
  const float* ln_b = (const float*)d_in[11];
  float* out = (float*)d_out;

  char* ws = (char*)d_ws;
  size_t off = 0;
  auto take = [&](size_t bytes) -> void* {
    void* p = ws + off;
    off += (bytes + 511) & ~((size_t)511);
    return p;
  };
  const size_t NH = (size_t)NROWS * HID;
  size_t zbase = off;
  float* kv   = (float*)take((size_t)HID * HID * 4);
  float* ksum = (float*)take(HID * 4);
  float* scal = (float*)take(16 * 4);
  size_t zbytes = off - zbase;
  unsigned short* kvbT = (unsigned short*)take((size_t)HID * HID * 2);
  unsigned short* fcwT = (unsigned short*)take((size_t)DIN * HID * 2);
  unsigned short* wqT  = (unsigned short*)take((size_t)NLAYERS * HID * HID * 2);
  unsigned short* wkT  = (unsigned short*)take((size_t)NLAYERS * HID * HID * 2);
  unsigned short* wvT  = (unsigned short*)take((size_t)NLAYERS * HID * HID * 2);
  float*          hA   = (float*)take(NH * 4);
  unsigned short* hbf  = (unsigned short*)take(NH * 2);
  unsigned short* qbf  = (unsigned short*)take(NH * 2);
  unsigned short* kT   = (unsigned short*)take(NH * 2);  // transposed [HID][NROWS]
  unsigned short* vbf  = (unsigned short*)take(NH * 2);  // row-major  [NROWS][HID]
  unsigned short* vT   = (unsigned short*)take(NH * 2);  // transposed [HID][NROWS]

  const int nblocksRows = NROWS / 32;  // 3125

  k_convert<<<(DIN * HID + 255) / 256, 256, 0, stream>>>(fc_w, Wq, Wk, Wv, fcwT, wqT, wkT, wvT);
  k_fc_ln_relu<<<nblocksRows, 256, 0, stream>>>(x, fcwT, fc_b, ln_g, ln_b, hA, hbf);

  for (int l = 0; l < NLAYERS; ++l) {
    hipMemsetAsync(kv, 0, zbytes, stream);
    k_qkv<<<nblocksRows, 256, 0, stream>>>(hbf, wqT + (size_t)l * HID * HID, bq + (size_t)l * HID,
                                           qbf, (unsigned short*)nullptr, scal, 0, ksum, 0);
    k_qkv<<<nblocksRows, 256, 0, stream>>>(hbf, wkT + (size_t)l * HID * HID, bk + (size_t)l * HID,
                                           (unsigned short*)nullptr, kT, scal, 1, ksum, 1);
    k_qkv<<<nblocksRows, 256, 0, stream>>>(hbf, wvT + (size_t)l * HID * HID, bv + (size_t)l * HID,
                                           vbf, vT, scal, -1, ksum, 0);
    dim3 g2(4, 64);
    k_kv<<<g2, 256, 0, stream>>>(kT, vT, kv, NROWS, 1568);
    k_kv_scale<<<(HID * HID) / 256, 256, 0, stream>>>(kv, scal, kvbT);
    const bool last = (l == NLAYERS - 1);
    k_attn<<<nblocksRows, 256, 0, stream>>>(qbf, kvbT, ksum, scal, vbf, hA,
                                            ln_g + (size_t)(l + 1) * HID,
                                            ln_b + (size_t)(l + 1) * HID,
                                            last ? out : hA, last ? (unsigned short*)nullptr : hbf);
  }
}
